// SupportMemoryAttentionSummaryV2_15934328668505
// MI455X (gfx1250) — matrix operand order repaired and checked by reading, not compiled, not run on hardware
//
#include <hip/hip_runtime.h>

// ---------------------------------------------------------------------------
// Fused SupportMemoryAttentionSummary for MI455X (gfx1250, wave32, WMMA).
//
//   LN(support) -> x@Wm^T+bm -> K=h@Wk^T+bk, V=h@Wv^T+bv
//   single-query-per-head attention (online softmax, split over M)
//   -> summary @ Wo^T + bo
//
// Heavy GEMMs on v_wmma_f32_16x16x32_f16, f32 accumulation.
// 64-row M-tiles (128 KB LDS) halve L2 weight traffic; 16 waves/WG keep
// per-wave pressure at 8 accumulator tiles; B operands double-buffered in
// registers so WMMAs wait with loadcnt<=2 instead of 0.
// ---------------------------------------------------------------------------

typedef __attribute__((ext_vector_type(16))) _Float16       v16h;
typedef __attribute__((ext_vector_type(8)))  float          v8f;
typedef __attribute__((ext_vector_type(8)))  unsigned short ushort8;

#define LATENT   512
#define HIDDEN   512
#define NHEAD    8
#define BB       128
#define MM       2048
#define SPLITS   8
#define ROWS_PER_SPLIT (MM / SPLITS)      // 256
#define MT       64                       // rows per tile
#define TILES    (ROWS_PER_SPLIT / MT)    // 4
#define NWAVES   16                       // 512 threads
#define PARTS_PER_BH (SPLITS * 2)         // 2 row-half partials per split
#define PART_STRIDE 66                    // [max, sum, acc0..63]

union ABU  { v16h v; ushort8 h[2]; };
union F16U { float4 v[4]; float f[16]; };

__device__ __forceinline__ unsigned short f2h_bits(float f) {
  return __builtin_bit_cast(unsigned short, (_Float16)f);
}
__device__ __forceinline__ float h2f_bits(unsigned short u) {
  return (float)__builtin_bit_cast(_Float16, u);
}
__device__ __forceinline__ float wred_sum(float v) {
  #pragma unroll
  for (int o = 16; o > 0; o >>= 1) v += __shfl_xor(v, o, 32);
  return v;
}
__device__ __forceinline__ float wred_max(float v) {
  #pragma unroll
  for (int o = 16; o > 0; o >>= 1) v = fmaxf(v, __shfl_xor(v, o, 32));
  return v;
}

// XOR-swizzled LDS tile: element (row, k) lives at
//   row*512 + (((k>>3) ^ (row&15)) << 3) + (k&7)      (16B chunk granularity)
// -> 16 rows at the same k hit 16 distinct banksets (conflict-free b128 loads).
__device__ __forceinline__ ushort8 lds_ld8(const unsigned short* s, int row, int k) {
  const int chunk = (k >> 3) ^ (row & 15);
  return *(const ushort8*)(s + (row << 9) + (chunk << 3));
}

// 64x512 output slab of  D = A[64x512](LDS,f16) x W^T[512x512](global,f16),
// split over 16 waves: wave owns row-tile `rowBase` (0/16/32/48) and
// col-group cg (0..3) -> 8 col-tiles = cols cg*128 .. +127.
// A layout (16x32, contiguous k order chosen here, not ISA table order): lane<16 -> K 0-15 ; lane>=16 -> K 16-31.
// B layout (32x16, same contiguous k order as A, not ISA table order): lane<16 -> K 0-15 ; lane>=16 -> K 16-31; col = lane&15.
// W[n][k] row-major is exactly K-contiguous per column: 2 b128 loads per operand.
// B operands are double-buffered (b0/b1) so each WMMA only needs the other
// buffer's loads outstanding -> graded s_wait_loadcnt, better XDL overlap.
__device__ __forceinline__ void gemm_64x512x512(const unsigned short* __restrict__ Wg,
                                                const unsigned short* src,
                                                v8f acc[8], int lane, int rowBase, int cg) {
  const v8f z = {0.f, 0.f, 0.f, 0.f, 0.f, 0.f, 0.f, 0.f};
  #pragma unroll
  for (int t = 0; t < 8; ++t) acc[t] = z;
  const int arow = rowBase + (lane & 15);
  const int alo  = (lane < 16) ? 0 : 16;
  const int blo  = (lane < 16) ? 0 : 16;
  const unsigned short* wbase = Wg + (cg * 128 + (lane & 15)) * 512 + blo;
  for (int ks = 0; ks < 512; ks += 32) {
    ABU a;
    a.h[0] = lds_ld8(src, arow, ks + alo);
    a.h[1] = lds_ld8(src, arow, ks + alo + 8);
    const unsigned short* wk = wbase + ks;
    ABU b0, b1;
    b0.h[0] = ((const ushort8*)wk)[0];
    b0.h[1] = ((const ushort8*)wk)[1];
    #pragma unroll
    for (int t = 0; t < 8; t += 2) {
      const ushort8* wp1 = (const ushort8*)(wk + ((t + 1) << 13));
      b1.h[0] = wp1[0];
      b1.h[1] = wp1[1];
      acc[t] = __builtin_amdgcn_wmma_f32_16x16x32_f16(
          /*neg_a=*/false, a.v, /*neg_b=*/false, b0.v,
          /*c_mod=*/(short)0, acc[t], /*reuse_a=*/false, /*reuse_b=*/false);
      if (t + 2 < 8) {
        const ushort8* wp2 = (const ushort8*)(wk + ((t + 2) << 13));
        b0.h[0] = wp2[0];
        b0.h[1] = wp2[1];
      }
      acc[t + 1] = __builtin_amdgcn_wmma_f32_16x16x32_f16(
          /*neg_a=*/false, a.v, /*neg_b=*/false, b1.v,
          /*c_mod=*/(short)0, acc[t + 1], /*reuse_a=*/false, /*reuse_b=*/false);
    }
  }
}

// C/D layout (ISA): VGPR r, lane<16 -> row rowBase+r, lane>=16 -> row rowBase+8+r,
// column = lane&15 within the 16-wide col tile.  +bias, f32->f16, swizzled LDS store.
__device__ __forceinline__ void store_tile(unsigned short* dst, v8f acc, float bias,
                                           int rowBase, int col, int lane) {
  const int r0 = rowBase + ((lane < 16) ? 0 : 8);
  #pragma unroll
  for (int r = 0; r < 8; ++r) {
    const int row = r0 + r;
    const int idx = (row << 9) + ((((col >> 3) ^ (row & 15)) << 3)) + (col & 7);
    dst[idx] = f2h_bits(acc[r] + bias);
  }
}

__global__ __launch_bounds__(512, 1)
void fused_mem_attn(const float* __restrict__ support,
                    const float* __restrict__ ln_w, const float* __restrict__ ln_b,
                    const unsigned short* __restrict__ Wm, const float* __restrict__ bm,
                    const unsigned short* __restrict__ Wk, const float* __restrict__ bk,
                    const unsigned short* __restrict__ Wv, const float* __restrict__ bv,
                    const float* __restrict__ qv, float* __restrict__ part) {
  __shared__ __align__(16) unsigned short bufA[MT * 512];  // LN(x) f16, later K f16 (64 KB)
  __shared__ __align__(16) unsigned short bufH[MT * 512];  // hidden f16, later V f16 (64 KB)
  const int sIdx = blockIdx.x;
  const int b    = blockIdx.y;
  const int wave = threadIdx.x >> 5;
  const int lane = threadIdx.x & 31;
  const int rowBase = (wave & 3) << 4;   // GEMM row-tile (0/16/32/48)
  const int cg      = wave >> 2;         // GEMM col-group (0..3), 8 col-tiles each
  const int h       = wave & 7;          // attention head for this wave
  const int half    = wave >> 3;         // which 32 of the 64 tile rows

  // Per-lane LN gamma/beta for columns lane*16 .. lane*16+15 (hoisted).
  F16U gw, gb;
  {
    const float4* p = (const float4*)(ln_w + lane * 16);
    gw.v[0] = p[0]; gw.v[1] = p[1]; gw.v[2] = p[2]; gw.v[3] = p[3];
    const float4* q = (const float4*)(ln_b + lane * 16);
    gb.v[0] = q[0]; gb.v[1] = q[1]; gb.v[2] = q[2]; gb.v[3] = q[3];
  }
  // Query for head h: lane holds d=lane and d=lane+32.
  const float q0 = qv[b * 512 + h * 64 + lane];
  const float q1 = qv[b * 512 + h * 64 + 32 + lane];

  // Online-softmax state (per wave: head h over its 32 rows of each tile).
  float mh = -1e30f, lh = 0.f, a0 = 0.f, a1 = 0.f;

  const int mStart = sIdx * ROWS_PER_SPLIT;
  for (int tile = 0; tile < TILES; ++tile) {
    const int m0 = mStart + tile * MT;
    __syncthreads();  // prior-tile attention reads done before we overwrite LDS

    // ---- LayerNorm: 16 waves x 4 rows; lane covers 16 of 512 columns ----
    #pragma unroll
    for (int i = 0; i < 4; ++i) {
      const int rowT = wave * 4 + i;
      const float* rp = support + (((size_t)b * MM + (size_t)(m0 + rowT)) << 9) + lane * 16;
      if (tile + 1 < TILES) __builtin_prefetch(rp + MT * 512, 0, 0);
      F16U x;
      const float4* gp = (const float4*)rp;
      x.v[0] = gp[0]; x.v[1] = gp[1]; x.v[2] = gp[2]; x.v[3] = gp[3];
      float s = 0.f, sq = 0.f;
      #pragma unroll
      for (int j = 0; j < 16; ++j) { s += x.f[j]; sq += x.f[j] * x.f[j]; }
      s  = wred_sum(s);
      sq = wred_sum(sq);
      const float mean = s * (1.f / 512.f);
      const float rstd = rsqrtf(sq * (1.f / 512.f) - mean * mean + 1e-5f);
      ushort8 o0, o1;
      #pragma unroll
      for (int j = 0; j < 8; ++j)
        o0[j] = f2h_bits((x.f[j] - mean) * rstd * gw.f[j] + gb.f[j]);
      #pragma unroll
      for (int j = 0; j < 8; ++j)
        o1[j] = f2h_bits((x.f[8 + j] - mean) * rstd * gw.f[8 + j] + gb.f[8 + j]);
      const int c0 = ((lane * 2)     ^ (rowT & 15)) << 3;
      const int c1 = ((lane * 2 + 1) ^ (rowT & 15)) << 3;
      *(ushort8*)(bufA + (rowT << 9) + c0) = o0;
      *(ushort8*)(bufA + (rowT << 9) + c1) = o1;
    }
    __syncthreads();

    v8f acc[8];
    // ---- GEMM1: H = LN(x) @ Wm^T + bm  -> bufH (f16) ----
    gemm_64x512x512(Wm, bufA, acc, lane, rowBase, cg);
    #pragma unroll
    for (int t = 0; t < 8; ++t) {
      const int col = cg * 128 + (t << 4) + (lane & 15);
      store_tile(bufH, acc[t], bm[col], rowBase, col, lane);
    }
    __syncthreads();

    // ---- GEMM2: K = H @ Wk^T + bk -> bufA (safe: all bufA reads done) ----
    gemm_64x512x512(Wk, bufH, acc, lane, rowBase, cg);
    #pragma unroll
    for (int t = 0; t < 8; ++t) {
      const int col = cg * 128 + (t << 4) + (lane & 15);
      store_tile(bufA, acc[t], bk[col], rowBase, col, lane);
    }

    // ---- GEMM3: V = H @ Wv^T + bv -> regs, then bufH after barrier ----
    gemm_64x512x512(Wv, bufH, acc, lane, rowBase, cg);
    __syncthreads();  // all bufH reads complete before overwrite
    #pragma unroll
    for (int t = 0; t < 8; ++t) {
      const int col = cg * 128 + (t << 4) + (lane & 15);
      store_tile(bufH, acc[t], bv[col], rowBase, col, lane);
    }
    __syncthreads();

    // ---- Attention: wave covers head h, rows half*32 + lane ----
    float logit = 0.f;
    {
      const int row = half * 32 + lane;
      #pragma unroll
      for (int cc = 0; cc < 8; ++cc) {
        const int chunk = (h * 8 + cc) ^ (row & 15);
        const ushort8 kv = *(const ushort8*)(bufA + (row << 9) + (chunk << 3));
        #pragma unroll
        for (int j = 0; j < 8; ++j) {
          const int d = cc * 8 + j;
          const float qd = (d < 32) ? __shfl(q0, d, 32) : __shfl(q1, d - 32, 32);
          logit = fmaf(qd, h2f_bits(kv[j]), logit);
        }
      }
      logit *= 0.125f;  // HEAD_DIM^-0.5
    }
    const float tmax = wred_max(logit);
    const float newm = fmaxf(mh, tmax);
    const float corr = __expf(mh - newm);
    const float p    = __expf(logit - newm);
    const float tsum = wred_sum(p);
    lh = lh * corr + tsum;
    mh = newm;
    a0 *= corr;
    a1 *= corr;
    const int cA = h * 64 + lane, cB = cA + 32;
    #pragma unroll 4
    for (int r = 0; r < 32; ++r) {
      const float pr = __shfl(p, r, 32);
      const int row = half * 32 + r;
      const int i0 = (row << 9) + ((((cA >> 3) ^ (row & 15)) << 3)) + (cA & 7);
      const int i1 = (row << 9) + ((((cB >> 3) ^ (row & 15)) << 3)) + (cB & 7);
      a0 = fmaf(pr, h2f_bits(bufH[i0]), a0);
      a1 = fmaf(pr, h2f_bits(bufH[i1]), a1);
    }
  }

  // Partial slot: (b, split*2 + half, head)
  float* pp = part + (size_t)(((b * SPLITS + sIdx) * 2 + half) * NHEAD + h) * PART_STRIDE;
  if (lane == 0) { pp[0] = mh; pp[1] = lh; }
  pp[2 + lane]  = a0;
  pp[34 + lane] = a1;
}

// Merge the PARTS_PER_BH partial softmax states per (b, head) -> normalized summary.
__global__ void combine_partials(const float* __restrict__ part, float* __restrict__ summary) {
  const int bh = blockIdx.x;  // B*NHEAD blocks of 32 threads
  const int b = bh >> 3, h = bh & 7;
  const int lane = threadIdx.x;
  float M = -1e30f;
  for (int s = 0; s < PARTS_PER_BH; ++s)
    M = fmaxf(M, part[(size_t)((b * PARTS_PER_BH + s) * NHEAD + h) * PART_STRIDE]);
  float L = 0.f, a0 = 0.f, a1 = 0.f;
  for (int s = 0; s < PARTS_PER_BH; ++s) {
    const float* pp = part + (size_t)((b * PARTS_PER_BH + s) * NHEAD + h) * PART_STRIDE;
    const float c = __expf(pp[0] - M);
    L  += c * pp[1];
    a0 += c * pp[2 + lane];
    a1 += c * pp[34 + lane];
  }
  const float inv = 1.f / L;
  summary[b * 512 + h * 64 + lane]      = a0 * inv;
  summary[b * 512 + h * 64 + 32 + lane] = a1 * inv;
}

// Y[r, n] = X[r, :512] . W[n, :512] + bias[n]   (tiny GEMMs: q-proj / out-proj)
__global__ __launch_bounds__(256)
void linear_512(const float* __restrict__ X, const float* __restrict__ W,
                const float* __restrict__ bias, float* __restrict__ Y) {
  const int t = blockIdx.x * blockDim.x + threadIdx.x;
  const int r = t >> 9, n = t & 511;
  const float4* x = (const float4*)(X + ((size_t)r << 9));
  const float4* w = (const float4*)(W + ((size_t)n << 9));
  float s = 0.f;
  #pragma unroll 8
  for (int i = 0; i < 128; ++i) {
    const float4 a = x[i], bvv = w[i];
    s = fmaf(a.x, bvv.x, s); s = fmaf(a.y, bvv.y, s);
    s = fmaf(a.z, bvv.z, s); s = fmaf(a.w, bvv.w, s);
  }
  Y[t] = s + bias[n];
}

__global__ void cvt_weights_f16(const float* __restrict__ Wm, const float* __restrict__ Wk,
                                const float* __restrict__ Wv, unsigned short* __restrict__ om,
                                unsigned short* __restrict__ ok, unsigned short* __restrict__ ov) {
  const int t = blockIdx.x * blockDim.x + threadIdx.x;  // exactly 512*512 threads
  om[t] = f2h_bits(Wm[t]);
  ok[t] = f2h_bits(Wk[t]);
  ov[t] = f2h_bits(Wv[t]);
}

extern "C" void kernel_launch(void* const* d_in, const int* in_sizes, int n_in,
                              void* d_out, int out_size, void* d_ws, size_t ws_size,
                              hipStream_t stream) {
  const float* query   = (const float*)d_in[0];
  const float* support = (const float*)d_in[1];
  const float* ln_w = (const float*)d_in[2];
  const float* ln_b = (const float*)d_in[3];
  const float* Wm = (const float*)d_in[4];
  const float* bm = (const float*)d_in[5];
  const float* Wq = (const float*)d_in[6];
  const float* bq = (const float*)d_in[7];
  const float* Wk = (const float*)d_in[8];
  const float* bk = (const float*)d_in[9];
  const float* Wv = (const float*)d_in[10];
  const float* bv = (const float*)d_in[11];
  const float* Wo = (const float*)d_in[12];
  const float* bo = (const float*)d_in[13];

  // Workspace layout (~6.2 MB total):
  char* ws = (char*)d_ws;
  float* ws_q    = (float*)(ws + 0);                  // 128*512 f32
  float* ws_sum  = (float*)(ws + 262144);             // 128*512 f32
  float* ws_part = (float*)(ws + 524288);             // 128*16*8*66 f32 (4,325,376 B)
  unsigned short* wm16 = (unsigned short*)(ws + 4849664);  // 512*512 f16
  unsigned short* wk16 = (unsigned short*)(ws + 5373952);
  unsigned short* wv16 = (unsigned short*)(ws + 5898240);

  cvt_weights_f16<<<1024, 256, 0, stream>>>(Wm, Wk, Wv, wm16, wk16, wv16);
  linear_512<<<256, 256, 0, stream>>>(query, Wq, bq, ws_q);
  fused_mem_attn<<<dim3(SPLITS, BB), 512, 0, stream>>>(
      support, ln_w, ln_b, wm16, bm, wk16, bk, wv16, bv, ws_q, ws_part);
  combine_partials<<<BB * NHEAD, 32, 0, stream>>>(ws_part, ws_sum);
  linear_512<<<256, 256, 0, stream>>>(ws_sum, Wo, bo, (float*)d_out);
}
